// AggregatorLayer_33406255628651
// MI455X (gfx1250) — matrix operand order repaired and checked by reading, not compiled, not run on hardware
//
#include <hip/hip_runtime.h>
#include <hip/hip_bf16.h>
#include <math.h>

// ---------------------------------------------------------------------------
// CDNA5 (gfx1250) wave32 WMMA building blocks: D = A(16x32 f16) x B(32x16 f16) + C(f32)
// ---------------------------------------------------------------------------
typedef __attribute__((ext_vector_type(16))) _Float16 v16h;
typedef __attribute__((ext_vector_type(8)))  float    v8f;

#define WMMA_F16(a, b, c) \
  __builtin_amdgcn_wmma_f32_16x16x32_f16(false, (a), false, (b), (short)0, (c), false, false)

// A tile 16x32 (MxK), row-major source with leading dim `ld` (halves).
// lanes 0-15 rows M=0..15 (g = lane>>4), halves i=0..15 -> K = g*16+i
__device__ __forceinline__ v16h load_a_tile(const _Float16* src, int ld) {
  int lane = threadIdx.x & 31;
  int m = lane & 15, g = lane >> 4;
  const _Float16* r = src + m * ld;
  v16h a;
#pragma unroll
  for (int i = 0; i < 16; ++i) a[i] = r[g * 16 + i];
  return a;
}
// B fragment from pre-swizzled weights: tile stored as 32 lanes x 16 contiguous halves
__device__ __forceinline__ v16h load_b_frag(const _Float16* Wsw, int tile) {
  int lane = threadIdx.x & 31;
  return *(const v16h*)(Wsw + ((size_t)tile * 32 + lane) * 16);
}
// B tile where logical B[k][n] = src[n*ld + k]  (contiguous along k: vectorizable)
__device__ __forceinline__ v16h load_bt_tile(const _Float16* src, int ld) {
  int lane = threadIdx.x & 31;
  int n = lane & 15, g = lane >> 4;
  const _Float16* r = src + n * ld + g * 16;
  v16h b;
#pragma unroll
  for (int i = 0; i < 16; ++i) b[i] = r[i];
  return b;
}
__device__ __forceinline__ float wred_sum(float v) {
#pragma unroll
  for (int m = 16; m >= 1; m >>= 1) v += __shfl_xor(v, m, 32);
  return v;
}
__device__ __forceinline__ float wred_max(float v) {
#pragma unroll
  for (int m = 16; m >= 1; m >>= 1) v = fmaxf(v, __shfl_xor(v, m, 32));
  return v;
}
__device__ __forceinline__ int reg3(int p) { return p < 24 ? 0 : (p < 28 ? 1 : 2); }

// ---------------------------------------------------------------------------
// Problem constants
// ---------------------------------------------------------------------------
#define TCLS 171
#define HID  128
#define AGD  256
#define TGD  256
#define RES  32
#define LPIX 1024
#define NWIN 16
#define PADL 256
#define NPOS 64
#define MROWS_SWIN (TCLS * LPIX)   // 175104
#define MROWS_CT   (NPOS * PADL)   // 16384

// ---------------------------------------------------------------------------
// Utility kernels
// ---------------------------------------------------------------------------
// Convert f32 weight (K x N, row-major) to f16 in WMMA B-fragment order:
// out[((ks*(N/16)+nt)*32 + lane)*16 + i] = W[(ks*32 + (lane>>4)*16 + i)*N + nt*16 + (lane&15)]
__global__ void k_swizzle_w(const float* __restrict__ W, _Float16* __restrict__ out,
                            int K, int N) {
  int idx = blockIdx.x * 256 + threadIdx.x;
  if (idx >= K * N) return;
  int i = idx & 15;
  int lane = (idx >> 4) & 31;
  int tile = idx >> 9;
  int ntiles = N >> 4;
  int ks = tile / ntiles, nt = tile - ks * ntiles;
  int kk = ks * 32 + (lane >> 4) * 16 + i;
  int nn = nt * 16 + (lane & 15);
  out[idx] = (_Float16)W[kk * N + nn];
}

// x input (B,C,T,H,W) -> xws (T, pix, C)
__global__ void k_transpose_in(const float* __restrict__ xin, float* __restrict__ xws) {
  int i = blockIdx.x * 256 + threadIdx.x;
  if (i >= HID * TCLS * LPIX) return;
  int c = i / (TCLS * LPIX);
  int r = i - c * (TCLS * LPIX);        // r = t*1024 + pix
  xws[r * HID + c] = xin[i];
}

// LayerNorm appearance guidance per pixel over AG=256 -> f16 (pix, 256)
__global__ __launch_bounds__(256) void k_guid_ln(const float* __restrict__ ag,
                                                 const float* __restrict__ g,
                                                 const float* __restrict__ b,
                                                 _Float16* __restrict__ gnh) {
  int wv = threadIdx.x >> 5, lane = threadIdx.x & 31;
  int pix = blockIdx.x * 8 + wv;
  float v[8], s = 0.f, s2 = 0.f;
#pragma unroll
  for (int i = 0; i < 8; ++i) {
    int c = lane + 32 * i;
    v[i] = ag[c * LPIX + pix];
    s += v[i]; s2 += v[i] * v[i];
  }
  s = wred_sum(s); s2 = wred_sum(s2);
  float mean = s * (1.f / 256.f);
  float rs = rsqrtf(s2 * (1.f / 256.f) - mean * mean + 1e-5f);
#pragma unroll
  for (int i = 0; i < 8; ++i) {
    int c = lane + 32 * i;
    gnh[pix * 256 + c] = (_Float16)((v[i] - mean) * rs * g[c] + b[c]);
  }
}

// ---------------------------------------------------------------------------
// Fused Swin window attention (one workgroup = one 8x8 window of one class-slice)
// LDS: wt 64x384 f16 | qt/kt 64x128 f16 | vtT 128x64 f16 | st 64x64 f32 | at 64x64 f16 | ot 64x128 f16
// ---------------------------------------------------------------------------
__global__ __launch_bounds__(256) void k_swin_attn(
    float* __restrict__ x, const _Float16* __restrict__ gnh,
    const float* __restrict__ n1g, const float* __restrict__ n1b,
    const _Float16* __restrict__ qw, const float* __restrict__ qb,
    const _Float16* __restrict__ kw, const float* __restrict__ kb,
    const _Float16* __restrict__ vw, const float* __restrict__ vb,
    const _Float16* __restrict__ pw, const float* __restrict__ pb,
    int shift) {
  extern __shared__ char smem[];
  _Float16* wt  = (_Float16*)smem;               // 64*384
  _Float16* qt  = wt + 64 * 384;                 // 64*128
  _Float16* kt  = qt + 64 * 128;                 // 64*128
  _Float16* vtT = kt + 64 * 128;                 // 128*64 (transposed: [c][token])
  float*    st  = (float*)(vtT + 128 * 64);      // 64*64 f32
  _Float16* at  = (_Float16*)(st + 64 * 64);     // 64*64
  _Float16* ot  = at + 64 * 64;                  // 64*128

  int bt = blockIdx.x >> 4;
  int win = blockIdx.x & 15;
  int wh = win >> 2, ww = win & 3;
  int wv = threadIdx.x >> 5, lane = threadIdx.x & 31;

  // ---- stage: LN(x) || guidance, with shift-roll gather ----
  for (int rr = 0; rr < 8; ++rr) {
    int n = wv * 8 + rr;
    int py = wh * 8 + (n >> 3), px = ww * 8 + (n & 7);
    int pix = ((py + shift) & 31) * RES + ((px + shift) & 31);
    const float* xr = x + (bt * LPIX + pix) * HID;
    float v0[4], s = 0.f, s2 = 0.f;
#pragma unroll
    for (int i = 0; i < 4; ++i) { float vv = xr[lane * 4 + i]; v0[i] = vv; s += vv; s2 += vv * vv; }
    s = wred_sum(s); s2 = wred_sum(s2);
    float mean = s * (1.f / 128.f);
    float rs = rsqrtf(s2 * (1.f / 128.f) - mean * mean + 1e-5f);
#pragma unroll
    for (int i = 0; i < 4; ++i) {
      int c = lane * 4 + i;
      wt[n * 384 + c] = (_Float16)((v0[i] - mean) * rs * n1g[c] + n1b[c]);
    }
    const _Float16* gr = gnh + pix * 256;
#pragma unroll
    for (int i = 0; i < 8; ++i) wt[n * 384 + 128 + lane + 32 * i] = gr[lane + 32 * i];
  }
  __syncthreads();

  // ---- q/k (64x384x128) and v (64x128x128), shared A fragments ----
  const float qscale = 0.17677669529663687f;  // 32^-0.5
  for (int t = wv; t < 32; t += 8) {
    int mt = t >> 3, nt = t & 7;
    v8f aq = {}, ak = {}, av = {};
    for (int ks = 0; ks < 12; ++ks) {
      v16h a = load_a_tile(wt + mt * 16 * 384 + ks * 32, 384);
      aq = WMMA_F16(a, load_b_frag(qw, ks * 8 + nt), aq);
      ak = WMMA_F16(a, load_b_frag(kw, ks * 8 + nt), ak);
      if (ks < 4) av = WMMA_F16(a, load_b_frag(vw, ks * 8 + nt), av);
    }
    int nl = lane & 15, g = lane >> 4;
#pragma unroll
    for (int r = 0; r < 8; ++r) {
      int m = mt * 16 + r + 8 * g, c = nt * 16 + nl;
      qt[m * HID + c]  = (_Float16)((aq[r] + qb[c]) * qscale);
      kt[m * HID + c]  = (_Float16)(ak[r] + kb[c]);
      vtT[c * 64 + m]  = (_Float16)(av[r] + vb[c]);   // transposed store for AV B-operand
    }
  }
  __syncthreads();

  // ---- per-head attention ----
  for (int h = 0; h < 4; ++h) {
    int ho = h * 32;
    for (int t = wv; t < 16; t += 8) {                 // S = q kT (K=32: one WMMA)
      int mt = t >> 2, nt = t & 3;
      v16h a = load_a_tile(qt + mt * 16 * HID + ho, HID);
      v16h b = load_bt_tile(kt + nt * 16 * HID + ho, HID);  // contiguous along channels
      v8f acc = {};
      acc = WMMA_F16(a, b, acc);
      int nl = lane & 15, g = lane >> 4;
#pragma unroll
      for (int r = 0; r < 8; ++r) {
        int m = mt * 16 + r + 8 * g, nn = nt * 16 + nl;
        float val = acc[r];
        if (shift) {
          int cm = reg3(wh * 8 + (m >> 3)) * 3 + reg3(ww * 8 + (m & 7));
          int cn = reg3(wh * 8 + (nn >> 3)) * 3 + reg3(ww * 8 + (nn & 7));
          if (cm != cn) val -= 100.f;
        }
        st[m * 64 + nn] = val;
      }
    }
    __syncthreads();
    for (int rr = 0; rr < 8; ++rr) {                   // softmax rows (wave per row)
      int m = wv * 8 + rr;
      float v0 = st[m * 64 + lane], v1 = st[m * 64 + 32 + lane];
      float mx = wred_max(fmaxf(v0, v1));
      float e0 = __expf(v0 - mx), e1 = __expf(v1 - mx);
      float ss = wred_sum(e0 + e1);
      float inv = 1.f / ss;
      at[m * 64 + lane]      = (_Float16)(e0 * inv);
      at[m * 64 + 32 + lane] = (_Float16)(e1 * inv);
    }
    __syncthreads();
    {                                                  // O_h = attn @ v_h (K=64)
      int mt = wv >> 1, nt = wv & 1;
      v8f acc = {};
      for (int ks = 0; ks < 2; ++ks) {
        v16h a = load_a_tile(at + mt * 16 * 64 + ks * 32, 64);
        // B[k][n] = v[token=k][ho+nt*16+n] = vtT[(ho+nt*16+n)*64 + k]  (contiguous along k)
        v16h b = load_bt_tile(vtT + (ho + nt * 16) * 64 + ks * 32, 64);
        acc = WMMA_F16(a, b, acc);
      }
      int nl = lane & 15, g = lane >> 4;
#pragma unroll
      for (int r = 0; r < 8; ++r) {
        int m = mt * 16 + r + 8 * g, c = ho + nt * 16 + nl;
        ot[m * HID + c] = (_Float16)acc[r];
      }
    }
    __syncthreads();
  }

  // ---- proj (64x128x128) + residual scatter (reverse roll == gather pixel) ----
  for (int t = wv; t < 32; t += 8) {
    int mt = t >> 3, nt = t & 7;
    v8f acc = {};
    for (int ks = 0; ks < 4; ++ks) {
      v16h a = load_a_tile(ot + mt * 16 * HID + ks * 32, HID);
      acc = WMMA_F16(a, load_b_frag(pw, ks * 8 + nt), acc);
    }
    int nl = lane & 15, g = lane >> 4;
#pragma unroll
    for (int r = 0; r < 8; ++r) {
      int m = mt * 16 + r + 8 * g, c = nt * 16 + nl;
      int py = wh * 8 + (m >> 3), px = ww * 8 + (m & 7);
      int pix = ((py + shift) & 31) * RES + ((px + shift) & 31);
      float* xr = x + (bt * LPIX + pix) * HID;
      xr[c] = xr[c] + acc[r] + pb[c];
    }
  }
}

// ---------------------------------------------------------------------------
// Fused MLP: x += act(LN2(x) @ fc1 + b1) @ fc2 + b2   (64 rows per workgroup)
// act: 0 = exact gelu (Swin), 1 = relu (class transformer)
// ---------------------------------------------------------------------------
__global__ __launch_bounds__(256) void k_mlp(
    float* __restrict__ x,
    const float* __restrict__ n2g, const float* __restrict__ n2b,
    const _Float16* __restrict__ f1w, const float* __restrict__ f1b,
    const _Float16* __restrict__ f2w, const float* __restrict__ f2b,
    int act) {
  extern __shared__ char smem[];
  _Float16* xh = (_Float16*)smem;     // 64*128
  _Float16* t1 = xh + 64 * 128;       // 64*512
  int r0 = blockIdx.x * 64;
  int wv = threadIdx.x >> 5, lane = threadIdx.x & 31;

  for (int rr = 0; rr < 8; ++rr) {
    int m = wv * 8 + rr;
    const float* xr = x + (size_t)(r0 + m) * HID;
    float v0[4], s = 0.f, s2 = 0.f;
#pragma unroll
    for (int i = 0; i < 4; ++i) { float vv = xr[lane * 4 + i]; v0[i] = vv; s += vv; s2 += vv * vv; }
    s = wred_sum(s); s2 = wred_sum(s2);
    float mean = s * (1.f / 128.f);
    float rs = rsqrtf(s2 * (1.f / 128.f) - mean * mean + 1e-5f);
#pragma unroll
    for (int i = 0; i < 4; ++i) {
      int c = lane * 4 + i;
      xh[m * HID + c] = (_Float16)((v0[i] - mean) * rs * n2g[c] + n2b[c]);
    }
  }
  __syncthreads();

  for (int t = wv; t < 128; t += 8) {                 // fc1: 64x128x512 (ntiles=32)
    int mt = t >> 5, nt = t & 31;
    v8f acc = {};
    for (int ks = 0; ks < 4; ++ks) {
      v16h a = load_a_tile(xh + mt * 16 * HID + ks * 32, HID);
      acc = WMMA_F16(a, load_b_frag(f1w, ks * 32 + nt), acc);
    }
    int nl = lane & 15, g = lane >> 4;
#pragma unroll
    for (int r = 0; r < 8; ++r) {
      int m = mt * 16 + r + 8 * g, c = nt * 16 + nl;
      float v = acc[r] + f1b[c];
      v = act ? fmaxf(v, 0.f) : 0.5f * v * (1.f + erff(v * 0.70710678118654752f));
      t1[m * 512 + c] = (_Float16)v;
    }
  }
  __syncthreads();

  for (int t = wv; t < 32; t += 8) {                  // fc2: 64x512x128, residual
    int mt = t >> 3, nt = t & 7;
    v8f acc = {};
    for (int ks = 0; ks < 16; ++ks) {
      v16h a = load_a_tile(t1 + mt * 16 * 512 + ks * 32, 512);
      acc = WMMA_F16(a, load_b_frag(f2w, ks * 8 + nt), acc);
    }
    int nl = lane & 15, g = lane >> 4;
#pragma unroll
    for (int r = 0; r < 8; ++r) {
      int m = mt * 16 + r + 8 * g, c = nt * 16 + nl;
      float* xr = x + (size_t)(r0 + m) * HID;
      xr[c] = xr[c] + acc[r] + f2b[c];
    }
  }
}

// ---------------------------------------------------------------------------
// Class-transformer helpers
// ---------------------------------------------------------------------------
__global__ void k_pool_pad(const float* __restrict__ xws, const float* __restrict__ pad_tok,
                           float* __restrict__ xp) {
  int gr = blockIdx.x;                 // pos*256 + tt
  int pos = gr >> 8, tt = gr & 255;
  int hp = pos >> 3, wp = pos & 7;
  int c = threadIdx.x;                 // 128 threads
  float v;
  if (tt < TCLS) {
    float s = 0.f;
#pragma unroll
    for (int dy = 0; dy < 4; ++dy)
#pragma unroll
      for (int dx = 0; dx < 4; ++dx) {
        int pix = (hp * 4 + dy) * RES + wp * 4 + dx;
        s += xws[((size_t)tt * LPIX + pix) * HID + c];
      }
    v = s * (1.f / 16.f);
  } else v = pad_tok[c];
  xp[(size_t)gr * HID + c] = v;
}

// LN1(xp) -> hg[:,0:128], guidance (text or pad) -> hg[:,128:384]
__global__ __launch_bounds__(256) void k_ct_hg(const float* __restrict__ xp,
                                               const float* __restrict__ n1g, const float* __restrict__ n1b,
                                               const float* __restrict__ tg, const float* __restrict__ pad_guid,
                                               _Float16* __restrict__ hg) {
  int wv = threadIdx.x >> 5, lane = threadIdx.x & 31;
  int gr = blockIdx.x * 8 + wv;
  int tt = gr & 255;
  const float* xr = xp + (size_t)gr * HID;
  float v0[4], s = 0.f, s2 = 0.f;
#pragma unroll
  for (int i = 0; i < 4; ++i) { float vv = xr[lane * 4 + i]; v0[i] = vv; s += vv; s2 += vv * vv; }
  s = wred_sum(s); s2 = wred_sum(s2);
  float mean = s * (1.f / 128.f);
  float rs = rsqrtf(s2 * (1.f / 128.f) - mean * mean + 1e-5f);
#pragma unroll
  for (int i = 0; i < 4; ++i) {
    int c = lane * 4 + i;
    hg[(size_t)gr * 384 + c] = (_Float16)((v0[i] - mean) * rs * n1g[c] + n1b[c]);
  }
  const float* gsrc = (tt < TCLS) ? tg + tt * TGD : pad_guid;
#pragma unroll
  for (int i = 0; i < 8; ++i)
    hg[(size_t)gr * 384 + 128 + lane + 32 * i] = (_Float16)gsrc[lane + 32 * i];
}

// Generic WMMA GEMM: out(M x 128) = A(f16, lda) @ Wsw(K x 128, fragment-swizzled) + bias
__global__ __launch_bounds__(256) void k_gemm128(const _Float16* __restrict__ A, int lda, int K,
                                                 const _Float16* __restrict__ W,
                                                 const float* __restrict__ bias,
                                                 float* __restrict__ out) {
  extern __shared__ char smem[];
  _Float16* at = (_Float16*)smem;     // 64*K
  int r0 = blockIdx.x * 64;
  int total = 64 * K;
  for (int i = threadIdx.x; i < total; i += 256) {
    int m = i / K, kk = i - m * K;
    at[i] = A[(size_t)(r0 + m) * lda + kk];
  }
  __builtin_prefetch(W + (size_t)threadIdx.x * 512, 0, 1);   // warm L2/WGP$ for fragments
  __syncthreads();
  int wv = threadIdx.x >> 5, lane = threadIdx.x & 31;
  for (int t = wv; t < 32; t += 8) {
    int mt = t >> 3, nt = t & 7;
    v8f acc = {};
    for (int ks = 0; ks < K / 32; ++ks) {
      v16h a = load_a_tile(at + mt * 16 * K + ks * 32, K);
      acc = WMMA_F16(a, load_b_frag(W, ks * 8 + nt), acc);
    }
    int nl = lane & 15, g = lane >> 4;
#pragma unroll
    for (int r = 0; r < 8; ++r) {
      int m = mt * 16 + r + 8 * g, c = nt * 16 + nl;
      out[(size_t)(r0 + m) * HID + c] = acc[r] + bias[c];
    }
  }
}

// ELU linear attention per spatial position; xp += out
__global__ __launch_bounds__(256) void k_linattn(const float* __restrict__ qb_,
                                                 const float* __restrict__ kb_,
                                                 const float* __restrict__ vb_,
                                                 float* __restrict__ xp) {
  extern __shared__ char smem[];
  float* Kp = (float*)smem;            // 256*128
  float* Vs = Kp + 256 * 128;          // 256*128
  float* KV = Vs + 256 * 128;          // 4*32*32
  float* Ks = KV + 4096;               // 128
  int pos = blockIdx.x, tid = threadIdx.x;
  size_t base = ((size_t)pos * 256 + tid) * HID;
  const float* krow = kb_ + base;
  const float* vrow = vb_ + base;
#pragma unroll 4
  for (int c = 0; c < 128; ++c) {
    float kv = krow[c];
    Kp[tid * 128 + c] = kv > 0.f ? kv + 1.f : __expf(kv);   // elu(k)+1
    Vs[tid * 128 + c] = vrow[c] * (1.f / 256.f);
  }
  __syncthreads();
  if (tid < 128) {
    float s = 0.f;
    for (int t = 0; t < 256; ++t) s += Kp[t * 128 + tid];
    Ks[tid] = s;
  }
  for (int i = 0; i < 16; ++i) {
    int e = i * 256 + tid;
    int hd = e >> 5;                   // h*32+d channel
    int dv = e & 31;
    int cv = (hd & ~31) | dv;          // h*32+dv channel
    float s = 0.f;
    for (int t = 0; t < 256; ++t) s += Kp[t * 128 + hd] * Vs[t * 128 + cv];
    KV[e] = s;
  }
  __syncthreads();
  float qp[128];
  const float* qrow = qb_ + base;
#pragma unroll 4
  for (int c = 0; c < 128; ++c) { float q = qrow[c]; qp[c] = q > 0.f ? q + 1.f : __expf(q); }
  float* xr = xp + base;
  for (int h = 0; h < 4; ++h) {
    float z = 0.f;
    for (int d = 0; d < 32; ++d) z += qp[h * 32 + d] * Ks[h * 32 + d];
    z = 1.f / (z + 1e-6f);
    for (int dv = 0; dv < 32; ++dv) {
      float o = 0.f;
      for (int d = 0; d < 32; ++d) o += qp[h * 32 + d] * KV[((h * 32 + d) << 5) | dv];
      xr[h * 32 + dv] += o * z * 256.f;
    }
  }
}

// out(B,C,T,H,W) = xws + bilinear_ac(xp 8x8 -> 32x32)
__global__ void k_upsample_add(const float* __restrict__ xws, const float* __restrict__ xp,
                               float* __restrict__ out) {
  int i = blockIdx.x * 256 + threadIdx.x;
  if (i >= HID * TCLS * LPIX) return;
  int c = i / (TCLS * LPIX);
  int rem = i - c * (TCLS * LPIX);
  int t = rem >> 10, p = rem & 1023;
  int y = p >> 5, xq = p & 31;
  float fy = y * (7.f / 31.f), fx = xq * (7.f / 31.f);
  int y0 = (int)fy; if (y0 > 7) y0 = 7;
  int x0 = (int)fx; if (x0 > 7) x0 = 7;
  int y1 = y0 + 1 > 7 ? 7 : y0 + 1;
  int x1 = x0 + 1 > 7 ? 7 : x0 + 1;
  float ty = fy - y0, tx = fx - x0;
  #define SMP(yy, xx) xp[(((size_t)(yy) * 8 + (xx)) * 256 + t) * HID + c]
  float v = SMP(y0, x0) * (1.f - ty) * (1.f - tx) + SMP(y1, x0) * ty * (1.f - tx)
          + SMP(y0, x1) * (1.f - ty) * tx + SMP(y1, x1) * ty * tx;
  #undef SMP
  out[i] = xws[((size_t)t * LPIX + p) * HID + c] + v;
}

// ---------------------------------------------------------------------------
// Host-side orchestration
// ---------------------------------------------------------------------------
extern "C" void kernel_launch(void* const* d_in, const int* in_sizes, int n_in,
                              void* d_out, int out_size, void* d_ws, size_t ws_size,
                              hipStream_t stream) {
  // Input order: x, appearance_guidance, text_guidance, then params pytree-flattened
  // (alphabetical: blk1, blk2, ct, gn_b, gn_g; alphabetical leaf keys within each).
  const float* x_in = (const float*)d_in[0];
  const float* ag   = (const float*)d_in[1];
  const float* tg   = (const float*)d_in[2];
  const float *b_fc1b[2], *b_fc1w[2], *b_fc2b[2], *b_fc2w[2], *b_kb[2], *b_kw[2],
              *b_n1b[2], *b_n1g[2], *b_n2b[2], *b_n2g[2], *b_pb[2], *b_pw[2],
              *b_qb[2], *b_qw[2], *b_vb[2], *b_vw[2];
  for (int blk = 0; blk < 2; ++blk) {
    int o = 3 + blk * 16;
    b_fc1b[blk] = (const float*)d_in[o + 0];  b_fc1w[blk] = (const float*)d_in[o + 1];
    b_fc2b[blk] = (const float*)d_in[o + 2];  b_fc2w[blk] = (const float*)d_in[o + 3];
    b_kb[blk]   = (const float*)d_in[o + 4];  b_kw[blk]   = (const float*)d_in[o + 5];
    b_n1b[blk]  = (const float*)d_in[o + 6];  b_n1g[blk]  = (const float*)d_in[o + 7];
    b_n2b[blk]  = (const float*)d_in[o + 8];  b_n2g[blk]  = (const float*)d_in[o + 9];
    b_pb[blk]   = (const float*)d_in[o + 10]; b_pw[blk]   = (const float*)d_in[o + 11];
    b_qb[blk]   = (const float*)d_in[o + 12]; b_qw[blk]   = (const float*)d_in[o + 13];
    b_vb[blk]   = (const float*)d_in[o + 14]; b_vw[blk]   = (const float*)d_in[o + 15];
  }
  const float* ct_fc1b = (const float*)d_in[35];
  const float* ct_fc1w = (const float*)d_in[36];
  const float* ct_fc2b = (const float*)d_in[37];
  const float* ct_fc2w = (const float*)d_in[38];
  const float* ct_kb   = (const float*)d_in[39];
  const float* ct_kw   = (const float*)d_in[40];
  const float* ct_n1b  = (const float*)d_in[41];
  const float* ct_n1g  = (const float*)d_in[42];
  const float* ct_n2b  = (const float*)d_in[43];
  const float* ct_n2g  = (const float*)d_in[44];
  const float* ct_padg = (const float*)d_in[45];
  const float* ct_padt = (const float*)d_in[46];
  const float* ct_qb   = (const float*)d_in[47];
  const float* ct_qw   = (const float*)d_in[48];
  const float* ct_vb   = (const float*)d_in[49];
  const float* ct_vw   = (const float*)d_in[50];
  const float* gn_b    = (const float*)d_in[51];
  const float* gn_g    = (const float*)d_in[52];

  // Workspace carve-out
  char* ws = (char*)d_ws;
  size_t off = 0;
  auto ALLOC = [&](size_t bytes) { size_t o = off; off = (off + bytes + 255) & ~(size_t)255; return o; };
  float*    xws  = (float*)   (ws + ALLOC((size_t)MROWS_SWIN * HID * 4));   // 89.6 MB
  _Float16* gnh  = (_Float16*)(ws + ALLOC((size_t)LPIX * AGD * 2));
  _Float16* w_qw[2], *w_kw[2], *w_vw[2], *w_pw[2], *w_f1[2], *w_f2[2];
  for (int blk = 0; blk < 2; ++blk) {
    w_qw[blk] = (_Float16*)(ws + ALLOC(384 * 128 * 2));
    w_kw[blk] = (_Float16*)(ws + ALLOC(384 * 128 * 2));
    w_vw[blk] = (_Float16*)(ws + ALLOC(128 * 128 * 2));
    w_pw[blk] = (_Float16*)(ws + ALLOC(128 * 128 * 2));
    w_f1[blk] = (_Float16*)(ws + ALLOC(128 * 512 * 2));
    w_f2[blk] = (_Float16*)(ws + ALLOC(512 * 128 * 2));
  }
  _Float16* cw_q  = (_Float16*)(ws + ALLOC(384 * 128 * 2));
  _Float16* cw_k  = (_Float16*)(ws + ALLOC(384 * 128 * 2));
  _Float16* cw_v  = (_Float16*)(ws + ALLOC(128 * 128 * 2));
  _Float16* cw_f1 = (_Float16*)(ws + ALLOC(128 * 512 * 2));
  _Float16* cw_f2 = (_Float16*)(ws + ALLOC(512 * 128 * 2));
  float*    xp    = (float*)   (ws + ALLOC((size_t)MROWS_CT * HID * 4));
  _Float16* hgct  = (_Float16*)(ws + ALLOC((size_t)MROWS_CT * 384 * 2));
  float*    qbuf  = (float*)   (ws + ALLOC((size_t)MROWS_CT * HID * 4));
  float*    kbuf  = (float*)   (ws + ALLOC((size_t)MROWS_CT * HID * 4));
  float*    vbuf  = (float*)   (ws + ALLOC((size_t)MROWS_CT * HID * 4));

  auto swz = [&](const float* s, _Float16* d, int K, int N) {
    k_swizzle_w<<<(K * N + 255) / 256, 256, 0, stream>>>(s, d, K, N);
  };
  for (int blk = 0; blk < 2; ++blk) {
    swz(b_qw[blk], w_qw[blk], 384, 128);
    swz(b_kw[blk], w_kw[blk], 384, 128);
    swz(b_vw[blk], w_vw[blk], 128, 128);
    swz(b_pw[blk], w_pw[blk], 128, 128);
    swz(b_fc1w[blk], w_f1[blk], 128, 512);
    swz(b_fc2w[blk], w_f2[blk], 512, 128);
  }
  swz(ct_qw, cw_q, 384, 128);
  swz(ct_kw, cw_k, 384, 128);
  swz(ct_vw, cw_v, 128, 128);
  swz(ct_fc1w, cw_f1, 128, 512);
  swz(ct_fc2w, cw_f2, 512, 128);

  int nElem = HID * TCLS * LPIX;
  k_transpose_in<<<(nElem + 255) / 256, 256, 0, stream>>>(x_in, xws);
  k_guid_ln<<<LPIX / 8, 256, 0, stream>>>(ag, gn_g, gn_b, gnh);

  const size_t ATTN_LDS = (64 * 384 + 2 * 64 * 128 + 128 * 64 + 64 * 64 + 64 * 128) * 2
                        + 64 * 64 * 4;                  // 139264
  const size_t MLP_LDS  = (64 * 128 + 64 * 512) * 2;    // 81920
  int nwin_blocks = TCLS * NWIN;   // 2736

  for (int blk = 0; blk < 2; ++blk) {
    int shift = blk ? 4 : 0;
    k_swin_attn<<<nwin_blocks, 256, ATTN_LDS, stream>>>(
        xws, gnh, b_n1g[blk], b_n1b[blk],
        w_qw[blk], b_qb[blk], w_kw[blk], b_kb[blk],
        w_vw[blk], b_vb[blk], w_pw[blk], b_pb[blk], shift);
    k_mlp<<<MROWS_SWIN / 64, 256, MLP_LDS, stream>>>(
        xws, b_n2g[blk], b_n2b[blk], w_f1[blk], b_fc1b[blk], w_f2[blk], b_fc2b[blk], 0);
  }

  // ---- class transformer ----
  k_pool_pad<<<MROWS_CT, 128, 0, stream>>>(xws, ct_padt, xp);
  k_ct_hg<<<MROWS_CT / 8, 256, 0, stream>>>(xp, ct_n1g, ct_n1b, tg, ct_padg, hgct);
  k_gemm128<<<MROWS_CT / 64, 256, 64 * 384 * 2, stream>>>(hgct, 384, 384, cw_q, ct_qb, qbuf);
  k_gemm128<<<MROWS_CT / 64, 256, 64 * 384 * 2, stream>>>(hgct, 384, 384, cw_k, ct_kb, kbuf);
  k_gemm128<<<MROWS_CT / 64, 256, 64 * 128 * 2, stream>>>(hgct, 384, 128, cw_v, ct_vb, vbuf);
  const size_t LA_LDS = (2 * 256 * 128 + 4096 + 128) * 4;   // 279,040 B (< 320KB WGP LDS)
  k_linattn<<<NPOS, 256, LA_LDS, stream>>>(qbuf, kbuf, vbuf, xp);
  k_mlp<<<MROWS_CT / 64, 256, MLP_LDS, stream>>>(
      xp, ct_n2g, ct_n2b, cw_f1, ct_fc1b, cw_f2, ct_fc2b, 1);

  // ---- upsample + final residual into (B,C,T,H,W) output ----
  k_upsample_add<<<(nElem + 255) / 256, 256, 0, stream>>>(xws, xp, (float*)d_out);

  (void)in_sizes; (void)n_in; (void)out_size; (void)ws_size;
}
